// DkvmnMilktModel_3745211482558
// MI455X (gfx1250) — compile-verified, matrix operand order checked by reading, not run on hardware
//
#include <hip/hip_runtime.h>
#include <hip/hip_bf16.h>

typedef __attribute__((ext_vector_type(16))) _Float16 v16h;
typedef __attribute__((ext_vector_type(8)))  _Float16 v8h;
typedef __attribute__((ext_vector_type(8)))  float    v8f;
typedef __attribute__((ext_vector_type(4)))  float    v4f;

#define WMMA_F16(a, b, c) \
  __builtin_amdgcn_wmma_f32_16x16x32_f16(false, (a), false, (b), (short)0, (c), false, false)

constexpr int B_  = 32;
constexpr int T_  = 512;
constexpr int BT  = B_ * T_;       // 16384
constexpr int D_  = 128;
constexpr int M_  = 50;
constexpr int NQ  = 1000;

// f16 weight pack layout (elements): [Mk padded 64x128][We 128x128][Wa 128x128][Wf 128x256]
constexpr int MK_OFF = 0;
constexpr int WE_OFF = 64 * D_;            // 8192
constexpr int WA_OFF = WE_OFF + D_ * D_;   // 24576
constexpr int WF_OFF = WA_OFF + D_ * D_;   // 40960
constexpr int WH_TOT = WF_OFF + D_ * 2 * D_; // 73728 halves

__device__ __forceinline__ v8f zero8() {
  v8f z;
#pragma unroll
  for (int i = 0; i < 8; ++i) z[i] = 0.0f;
  return z;
}

// 16x32 f16 A/B fragment from f16 memory (LDS or global).
// p = row/col base + kbase + sel*8 ; chunks are 16B aligned.
__device__ __forceinline__ v16h load_frag_h(const _Float16* p) {
  v8h c0 = *(const v8h*)(p);
  v8h c1 = *(const v8h*)(p + 16);
  v16h f;
#pragma unroll
  for (int i = 0; i < 8; ++i) { f[i] = c0[i]; f[8 + i] = c1[i]; }
  return f;
}

// A-fragment from f32 row-major memory (convert to f16).
__device__ __forceinline__ v16h load_frag_f32(const float* p) {
  v4f c0 = *(const v4f*)(p);
  v4f c1 = *(const v4f*)(p + 4);
  v4f c2 = *(const v4f*)(p + 16);
  v4f c3 = *(const v4f*)(p + 20);
  v16h f;
#pragma unroll
  for (int i = 0; i < 4; ++i) {
    f[i]      = (_Float16)c0[i];
    f[4 + i]  = (_Float16)c1[i];
    f[8 + i]  = (_Float16)c2[i];
    f[12 + i] = (_Float16)c3[i];
  }
  return f;
}

// Async copy `nchunks` x 16B from global (f16 pack) into LDS, one wave32-lane
// stream per thread. ASYNCcnt path: global_load_async_to_lds_b128.
__device__ __forceinline__ void async_copy_to_lds(void* lds_dst, const void* gsrc,
                                                  int nchunks, int tid, int nthreads) {
  const unsigned int dstbase = (unsigned int)(size_t)lds_dst;  // flat->LDS offset (addr[31:0])
  const unsigned long long src = (unsigned long long)(size_t)gsrc;
  for (int i = tid; i < nchunks; i += nthreads) {
    const unsigned int off = (unsigned int)i * 16u;
    asm volatile("global_load_async_to_lds_b128 %0, %1, %2"
                 :: "v"(dstbase + off), "v"(off), "s"(src)
                 : "memory");
  }
  asm volatile("s_wait_asynccnt 0" ::: "memory");
}

// ---------------------------------------------------------------------------
// Kernel 0: one-time weight conversion f32 -> f16 pack (Mk zero-padded to 64).
// ---------------------------------------------------------------------------
__global__ __launch_bounds__(256) void wconv_kernel(
    const float* __restrict__ Mk, const float* __restrict__ We,
    const float* __restrict__ Wa, const float* __restrict__ Wf,
    _Float16* __restrict__ wh) {
  const int i = blockIdx.x * 256 + threadIdx.x;
  if (i >= WH_TOT) return;
  float v;
  if (i < WE_OFF) {
    const int row = i >> 7;                       // Mk, padded rows >= 50 are 0
    v = (row < M_) ? Mk[i] : 0.0f;
  } else if (i < WA_OFF) {
    v = We[i - WE_OFF];
  } else if (i < WF_OFF) {
    v = Wa[i - WA_OFF];
  } else {
    v = Wf[i - WF_OFF];
  }
  wh[i] = (_Float16)v;
}

// ---------------------------------------------------------------------------
// Phase 1: gather k,v; w = softmax(k Mk^T); e = sigmoid(v We^T + be);
//          a = tanh(v Wa^T + ba). One wave per 16-row tile, 4 waves/block.
// Dynamic LDS layout (131072 B):
//   [0,16K)   kh[4][16][128] f16      [16K,32K) vh f16
//   [32K,48K) lg[4][16][64]  f32      [48K,128K) f16 weights (Mk|We|Wa), async-staged
// ---------------------------------------------------------------------------
__global__ __launch_bounds__(128) void phase1_kernel(
    const int* __restrict__ q, const int* __restrict__ r,
    const float* __restrict__ Ek, const float* __restrict__ Ev,
    const _Float16* __restrict__ wh,
    const float* __restrict__ be_, const float* __restrict__ ba_,
    float* __restrict__ w_out, float* __restrict__ e_out,
    float* __restrict__ a_out, _Float16* __restrict__ kh_out) {
  extern __shared__ __align__(16) char smem[];
  _Float16* khl = (_Float16*)smem;                    // [4][16][128]
  _Float16* vhl = (_Float16*)(smem + 16384);          // [4][16][128]
  float*    lgl = (float*)(smem + 32768);             // [4][16][64]
  _Float16* mkl = (_Float16*)(smem + 49152);          // [64][128]
  _Float16* wel = (_Float16*)(smem + 49152 + 16384);  // [128][128]
  _Float16* wal = wel + D_ * D_;                      // [128][128]

  const int tid  = threadIdx.x;
  const int wv   = tid >> 5;
  const int lane = tid & 31;
  const int l15  = lane & 15;
  const int sel  = lane >> 4;
  const int rowbase = (blockIdx.x * 4 + wv) * 16;

  // Kick off async staging of Mk|We|Wa f16 pack (80 KB = 5120 chunks).
  async_copy_to_lds(mkl, wh, 5120, tid, 128);

  // Stage 16 rows of k and v as f16 in LDS; spill k (f16) for phase 3.
  for (int rr = 0; rr < 16; ++rr) {
    const int gi   = rowbase + rr;
    const int qi   = q[gi];
    const int ri   = r[gi];
    const int msk  = (ri != 2) ? 1 : 0;
    const int xi   = (qi + NQ * ri) * msk;
    const float* krow = Ek + (size_t)qi * D_;
    const float* vrow = Ev + (size_t)xi * D_;
    const int c0 = lane * 4;
#pragma unroll
    for (int j = 0; j < 4; ++j) {
      _Float16 kvh = (_Float16)krow[c0 + j];
      khl[(wv * 16 + rr) * D_ + c0 + j] = kvh;
      vhl[(wv * 16 + rr) * D_ + c0 + j] = (_Float16)vrow[c0 + j];
      kh_out[(size_t)gi * D_ + c0 + j] = kvh;
    }
  }
  __syncthreads();   // k/v staged AND per-wave async weight chunks complete

  // ---- logits = k @ Mk^T, N padded 50 -> 64 (padded cols are exactly 0) ----
  v16h ak[4];
#pragma unroll
  for (int kt = 0; kt < 4; ++kt)
    ak[kt] = load_frag_h(khl + (wv * 16 + l15) * D_ + kt * 32 + sel * 8);

  v8f accL[4];
#pragma unroll
  for (int nt = 0; nt < 4; ++nt) accL[nt] = zero8();
#pragma unroll
  for (int kt = 0; kt < 4; ++kt) {
#pragma unroll
    for (int nt = 0; nt < 4; ++nt) {
      v16h bfrag = load_frag_h(mkl + (nt * 16 + l15) * D_ + kt * 32 + sel * 8);
      accL[nt] = WMMA_F16(ak[kt], bfrag, accL[nt]);
    }
  }
#pragma unroll
  for (int nt = 0; nt < 4; ++nt)
#pragma unroll
    for (int j = 0; j < 8; ++j)
      lgl[(wv * 16 + j + 8 * sel) * 64 + nt * 16 + l15] = accL[nt][j];
  __syncthreads();

  // ---- per-row softmax over the 50 valid columns (lanes 0..15 own rows) ----
  if (lane < 16) {
    float* lrow = lgl + (wv * 16 + lane) * 64;
    float mx = -1e30f;
    for (int m = 0; m < M_; ++m) mx = fmaxf(mx, lrow[m]);
    float s = 0.0f;
    for (int m = 0; m < M_; ++m) {
      float ex = __expf(lrow[m] - mx);
      lrow[m] = ex;
      s += ex;
    }
    const float inv = 1.0f / s;
    float* wrow = w_out + (size_t)(rowbase + lane) * M_;
    for (int m = 0; m < M_; ++m) wrow[m] = lrow[m] * inv;
  }

  // ---- e = sigmoid(v We^T + be), a = tanh(v Wa^T + ba) ----
  v16h av[4];
#pragma unroll
  for (int kt = 0; kt < 4; ++kt)
    av[kt] = load_frag_h(vhl + (wv * 16 + l15) * D_ + kt * 32 + sel * 8);

#pragma unroll
  for (int nt = 0; nt < 8; ++nt) {
    v8f accE = zero8(), accA = zero8();
    const int col = nt * 16 + l15;
#pragma unroll
    for (int kt = 0; kt < 4; ++kt) {
      v16h bE = load_frag_h(wel + col * D_ + kt * 32 + sel * 8);
      accE = WMMA_F16(av[kt], bE, accE);
      v16h bA = load_frag_h(wal + col * D_ + kt * 32 + sel * 8);
      accA = WMMA_F16(av[kt], bA, accA);
    }
    const float beC = be_[col], baC = ba_[col];
#pragma unroll
    for (int j = 0; j < 8; ++j) {
      const int row = rowbase + j + 8 * sel;
      const float xe = accE[j] + beC;
      e_out[(size_t)row * D_ + col] = 1.0f / (1.0f + __expf(-xe));
      const float xa = accA[j] + baC;
      a_out[(size_t)row * D_ + col] = tanhf(xa);
    }
  }
}

// ---------------------------------------------------------------------------
// Phase 2: serial DKVMN scan over T (the latency-critical path).
// One block per batch. Thread d owns Mv column d: the 50 Mv values live in
// VGPRs (no LDS traffic). w[b,t,:] is block-uniform -> SMEM broadcast loads.
// No barriers at all; read uses PRE-update memory, matching the reference.
// ---------------------------------------------------------------------------
__global__ __launch_bounds__(128) void scan_kernel(
    const float* __restrict__ Mv0, const float* __restrict__ w,
    const float* __restrict__ e, const float* __restrict__ a,
    float* __restrict__ reads) {
  const int b = blockIdx.x;
  const int d = threadIdx.x;  // 0..127

  float mv[M_];
#pragma unroll
  for (int m = 0; m < M_; ++m) mv[m] = Mv0[m * D_ + d];

  for (int t = 0; t < T_; ++t) {
    const int base = b * T_ + t;
    const float* wrow = w + (size_t)base * M_;   // uniform address -> s_load
    const float ev = e[(size_t)base * D_ + d];
    const float av = a[(size_t)base * D_ + d];

    float acc = 0.0f;
#pragma unroll
    for (int m = 0; m < M_; ++m) {
      const float wm = wrow[m];
      acc += wm * mv[m];                              // read (pre-update)
      mv[m] = mv[m] * (1.0f - wm * ev) + wm * av;     // erase + add
    }
    reads[(size_t)base * D_ + d] = acc;
  }
}

// ---------------------------------------------------------------------------
// Phase 3: f = tanh([reads, k] @ Wf^T + bf) ; p = sigmoid(f . Wp + bp)
// Dynamic LDS (98304 B): [0,32K) fl[4][16][128] f32 ; [32K,96K) Wf f16 (async)
// ---------------------------------------------------------------------------
__global__ __launch_bounds__(128) void phase3_kernel(
    const float* __restrict__ reads, const _Float16* __restrict__ kh,
    const _Float16* __restrict__ wh,
    const float* __restrict__ bf_, const float* __restrict__ Wp,
    const float* __restrict__ bp_, float* __restrict__ p_out) {
  extern __shared__ __align__(16) char smem[];
  float*    fl  = (float*)smem;                  // [4][16][128]
  _Float16* wfl = (_Float16*)(smem + 32768);     // [128][256]

  const int tid  = threadIdx.x;
  const int wv   = tid >> 5;
  const int lane = tid & 31;
  const int l15  = lane & 15;
  const int sel  = lane >> 4;
  const int rowbase = (blockIdx.x * 4 + wv) * 16;

  // Async-stage Wf f16 (64 KB = 4096 chunks) into LDS.
  async_copy_to_lds(wfl, wh + WF_OFF, 4096, tid, 128);

  // A-fragments, K = 256: kt 0..3 <- reads (f32->f16), kt 4..7 <- k (f16).
  v16h afr[8];
  const float*    rrow = reads + (size_t)(rowbase + l15) * D_;
  const _Float16* krow = kh    + (size_t)(rowbase + l15) * D_;
#pragma unroll
  for (int kt = 0; kt < 4; ++kt)
    afr[kt] = load_frag_f32(rrow + kt * 32 + sel * 8);
#pragma unroll
  for (int kt = 0; kt < 4; ++kt)
    afr[4 + kt] = load_frag_h(krow + kt * 32 + sel * 8);

  __syncthreads();   // Wf staged (per-wave async waits done in async_copy_to_lds)

#pragma unroll
  for (int nt = 0; nt < 8; ++nt) {
    v8f acc = zero8();
    const int col = nt * 16 + l15;
#pragma unroll
    for (int kt = 0; kt < 8; ++kt) {
      v16h b = load_frag_h(wfl + col * (2 * D_) + kt * 32 + sel * 8);
      acc = WMMA_F16(afr[kt], b, acc);
    }
    const float bc = bf_[col];
#pragma unroll
    for (int j = 0; j < 8; ++j)
      fl[(wv * 16 + j + 8 * sel) * D_ + col] = tanhf(acc[j] + bc);
  }
  __syncthreads();

  // p = sigmoid(f . Wp + bp); lanes 0..15 each own one row.
  if (lane < 16) {
    float acc = bp_[0];
    const float* frow = fl + (wv * 16 + lane) * D_;
    for (int dd = 0; dd < D_; ++dd) acc += frow[dd] * Wp[dd];
    p_out[rowbase + lane] = 1.0f / (1.0f + __expf(-acc));
  }
}

// ---------------------------------------------------------------------------
extern "C" void kernel_launch(void* const* d_in, const int* in_sizes, int n_in,
                              void* d_out, int out_size, void* d_ws, size_t ws_size,
                              hipStream_t stream) {
  (void)in_sizes; (void)n_in; (void)out_size; (void)ws_size;
  const int*   q   = (const int*)d_in[0];
  const int*   r   = (const int*)d_in[1];
  const float* Ek  = (const float*)d_in[2];
  const float* Ev  = (const float*)d_in[3];
  const float* Mk  = (const float*)d_in[4];
  const float* Mv0 = (const float*)d_in[5];
  const float* We  = (const float*)d_in[6];
  const float* be_ = (const float*)d_in[7];
  const float* Wa  = (const float*)d_in[8];
  const float* ba_ = (const float*)d_in[9];
  const float* Wf  = (const float*)d_in[10];
  const float* bf_ = (const float*)d_in[11];
  const float* Wp  = (const float*)d_in[12];
  const float* bp_ = (const float*)d_in[13];
  float* out = (float*)d_out;

  // Scratch layout (all chunk sizes multiples of 16 B):
  char* ws = (char*)d_ws;
  _Float16* wh_s = (_Float16*)ws;      ws += (size_t)WH_TOT * sizeof(_Float16);  // 147 KB f16 weights
  float* w_s  = (float*)ws;            ws += (size_t)BT * M_ * sizeof(float);    // 3.28 MB
  float* e_s  = (float*)ws;            ws += (size_t)BT * D_ * sizeof(float);    // 8.39 MB
  float* a_s  = (float*)ws;            ws += (size_t)BT * D_ * sizeof(float);    // 8.39 MB
  float* rd_s = (float*)ws;            ws += (size_t)BT * D_ * sizeof(float);    // 8.39 MB
  _Float16* kh_s = (_Float16*)ws;      // BT*D f16 = 4.19 MB  (total ~32.9 MB)

  wconv_kernel<<<(WH_TOT + 255) / 256, 256, 0, stream>>>(Mk, We, Wa, Wf, wh_s);
  phase1_kernel<<<BT / 64, 128, 131072, stream>>>(q, r, Ek, Ev, wh_s, be_, ba_,
                                                  w_s, e_s, a_s, kh_s);
  scan_kernel<<<B_, 128, 0, stream>>>(Mv0, w_s, e_s, a_s, rd_s);
  phase3_kernel<<<BT / 64, 128, 98304, stream>>>(rd_s, kh_s, wh_s, bf_, Wp, bp_, out);
}
